// STAG_17566416240842
// MI455X (gfx1250) — compile-verified, matrix operand order checked by reading, not run on hardware
//
#include <hip/hip_runtime.h>
#include <hip/hip_bf16.h>

#define Nn  50000
#define Ee  150000
#define Gg  200
#define Dd  320
#define EHh 32
#define NTt 9
#define NBb 3

typedef __attribute__((ext_vector_type(16))) __bf16 v16bf;
typedef __attribute__((ext_vector_type(8)))  __bf16 v8bf;
typedef __attribute__((ext_vector_type(8)))  float  v8f;
typedef __attribute__((ext_vector_type(4)))  unsigned int u32x4;
typedef __attribute__((ext_vector_type(8)))  int    i32x8;
typedef __attribute__((ext_vector_type(4)))  int    i32x4;

__device__ __forceinline__ void atomicMaxF(float* addr, float val) {
    unsigned* ua = (unsigned*)addr;
    unsigned old = *ua;
    while (__uint_as_float(old) < val) {
        unsigned assumed = old;
        old = atomicCAS(ua, assumed, __float_as_uint(val));
        if (old == assumed) break;
    }
}

// ---------------- generic fills / copies ----------------
__global__ void fill_k(float* p, float v, long n) {
    long i = (long)blockIdx.x * blockDim.x + threadIdx.x;
    if (i < n) p[i] = v;
}
__global__ void fill2_k(float* a, float va, float* b, float vb, long n) {
    long i = (long)blockIdx.x * blockDim.x + threadIdx.x;
    if (i < n) { a[i] = va; b[i] = vb; }
}
__global__ void copy_k(float* dst, const float* src, long n) {
    long i = (long)blockIdx.x * blockDim.x + threadIdx.x;
    if (i < n) dst[i] = src[i];
}
__global__ void count_k(const int* batch, float* cnt) {
    int i = blockIdx.x * blockDim.x + threadIdx.x;
    if (i < Nn) atomicAdd(&cnt[batch[i]], 1.0f);
}

// ---------------- WMMA GEMM: C[MxNcol] = A[MxK] @ B[KxNcol] (+bias) ----------------
// Block = 128 threads (4 waves). Block tile 64(M) x 64(N); each wave owns a
// 16x64 strip (4 f32 accumulators sharing one A fragment per 32-K chunk).
// A tile staged by the Tensor Data Mover (f32, per-wave 16x32 subtile, HW
// zero-fill past M); B staged via global_load_b128 + v_cvt_pk_bf16_f32 into
// transposed bf16 LDS. Requires K % 32 == 0 (K in {32,320}).
#define BM 64
#define BN 64
#define LSTR 40   // padded bf16 stride for B (80B, 16B-aligned)

__global__ void __launch_bounds__(128)
gemm_wmma_k(const float* __restrict__ A, const float* __restrict__ B,
            const float* __restrict__ bias, float* __restrict__ C,
            int M, int K, int Ncol, int accum) {
    __shared__ __align__(16) float  lAf[BM * 32];    // [row][k] f32 (TDM dest)
    __shared__ __align__(16) __bf16 lB[BN * LSTR];   // [n][k] bf16 (transposed)
    const int tid  = threadIdx.x;
    const int lane = tid & 31;
    const int wv   = tid >> 5;            // wave -> M subtile 0..3
    const int half = lane >> 4;           // frag half (K grouping per ISA layout)
    const int lid  = lane & 15;
    const long mbase = (long)blockIdx.x * BM;
    const long mwv   = mbase + wv * 16;   // this wave's row base
    const int  nbase = blockIdx.y * BN;

    v8f acc[4];
#pragma unroll
    for (int c = 0; c < 4; ++c) acc[c] = (v8f){0.f,0.f,0.f,0.f,0.f,0.f,0.f,0.f};

    for (int k0 = 0; k0 < K; k0 += 32) {
        __syncthreads();   // protect LDS from previous iteration's readers

        // ---- A tile: TDM async copy, one 16x32 f32 subtile per wave ----
        if (lane == 0 && mwv < M) {
            unsigned long long ga = (unsigned long long)(const void*)(A + mwv * K + k0);
            unsigned ldsa = (unsigned)(unsigned long long)(uintptr_t)lAf
                          + (unsigned)(wv * 16 * 32 * 4);
            unsigned td0 = (unsigned)(K - k0);     // K extent for OOB
            unsigned td1 = (unsigned)(M - mwv);    // rows left -> HW zero-fill
            u32x4 g0;
            g0[0] = 1u;                                    // count=1 (valid user D#)
            g0[1] = ldsa;                                  // lds_addr
            g0[2] = (unsigned)(ga & 0xFFFFFFFFull);        // global_addr[31:0]
            g0[3] = (unsigned)((ga >> 32) & 0x1FFFFFFull)  // global_addr[56:32]
                  | 0x80000000u;                           // type=2 ("image")
            i32x8 g1;
            g1[0] = (int)(2u << 16);                       // data_size=4B
            g1[1] = (int)((td0 & 0xFFFFu) << 16);          // tensor_dim0[15:0]
            g1[2] = (int)((td0 >> 16) | ((td1 & 0xFFFFu) << 16));
            g1[3] = (int)((td1 >> 16) | (32u << 16));      // tile_dim0 = 32
            g1[4] = (int)16;                               // tile_dim1 = 16
            g1[5] = (int)(unsigned)K;                      // tensor_dim0_stride lo32
            g1[6] = 0;
            g1[7] = 0;
            i32x4 gz = {0, 0, 0, 0};
#if defined(__clang_major__) && __clang_major__ >= 23
            i32x8 gz8 = {0, 0, 0, 0, 0, 0, 0, 0};
            __builtin_amdgcn_tensor_load_to_lds(g0, g1, gz, gz, gz8, 0);
#else
            __builtin_amdgcn_tensor_load_to_lds(g0, g1, gz, gz, 0);
#endif
        }

        // ---- stage B tile 32x64 transposed to [n][k] (f32 -> bf16) ----
#pragma unroll
        for (int it = 0; it < 4; ++it) {
            int idx = tid + it * 128;                 // 0..511 float4 slots
            int kk  = idx >> 4;
            int n4  = (idx & 15) * 4;
            const float* bp = B + (long)(k0 + kk) * Ncol + nbase + n4;
            float4 v = make_float4(0.f, 0.f, 0.f, 0.f);
            if (nbase + BN <= Ncol) {
                v = *(const float4*)(bp);
            } else {
#pragma unroll
                for (int j = 0; j < 4; ++j)
                    if (nbase + n4 + j < Ncol) (&v.x)[j] = bp[j];
            }
            lB[(n4 + 0) * LSTR + kk] = (__bf16)v.x;
            lB[(n4 + 1) * LSTR + kk] = (__bf16)v.y;
            lB[(n4 + 2) * LSTR + kk] = (__bf16)v.z;
            lB[(n4 + 3) * LSTR + kk] = (__bf16)v.w;
        }

        // prefetch next B chunk (speculative, global_prefetch_b8)
        if (k0 + 32 < K)
            __builtin_prefetch(B + (long)(k0 + 32 + (tid >> 2)) * Ncol + nbase, 0, 1);

        __builtin_amdgcn_s_wait_tensorcnt(0);   // this wave's A subtile landed
        __syncthreads();                        // B tile visible block-wide

        // ---- A fragment from f32 LDS (4x ds_load_b128) + packed bf16 cvt ----
        const float* ap = &lAf[(wv * 16 + lid) * 32 + half * 8];
        float4 a0 = *(const float4*)(ap);
        float4 a1 = *(const float4*)(ap + 4);
        float4 a2 = *(const float4*)(ap + 16);
        float4 a3 = *(const float4*)(ap + 20);
        v16bf af;
        af[0]=(__bf16)a0.x; af[1]=(__bf16)a0.y; af[2]=(__bf16)a0.z; af[3]=(__bf16)a0.w;
        af[4]=(__bf16)a1.x; af[5]=(__bf16)a1.y; af[6]=(__bf16)a1.z; af[7]=(__bf16)a1.w;
        af[8]=(__bf16)a2.x; af[9]=(__bf16)a2.y; af[10]=(__bf16)a2.z; af[11]=(__bf16)a2.w;
        af[12]=(__bf16)a3.x; af[13]=(__bf16)a3.y; af[14]=(__bf16)a3.z; af[15]=(__bf16)a3.w;

#pragma unroll
        for (int c = 0; c < 4; ++c) {
            const __bf16* bp = &lB[(c * 16 + lid) * LSTR + half * 8];
            v8bf blo = *(const v8bf*)(bp);
            v8bf bhi = *(const v8bf*)(bp + 16);
            v16bf bf = __builtin_shufflevector(blo, bhi,
                           0,1,2,3,4,5,6,7,8,9,10,11,12,13,14,15);
            acc[c] = __builtin_amdgcn_wmma_f32_16x16x32_bf16(false, af, false, bf,
                                                             (short)0, acc[c], false, false);
        }
    }

    // ---- store: lane -> N column, vgpr v -> M row (+8 for upper half) ----
#pragma unroll
    for (int c = 0; c < 4; ++c) {
        int n = nbase + c * 16 + lid;
        if (n >= Ncol) continue;
        float bv = bias ? bias[n] : 0.f;
#pragma unroll
        for (int v = 0; v < 8; ++v) {
            long m = mbase + wv * 16 + half * 8 + v;
            if (m < M) {
                long idx = m * Ncol + n;
                float r = acc[c][v] + bv;
                if (accum) C[idx] += r; else C[idx] = r;
            }
        }
    }
}

// ---------------- edge MLP: new_ea = lrelu(Xa[src]+Xb[dst]+ea@W1c+b1) @ W2 + b2 ----------------
__global__ void edge_mlp_k(const float* __restrict__ xa, const float* __restrict__ xb,
                           const int* __restrict__ EI, const float* __restrict__ EA,
                           const float* __restrict__ W1, const float* __restrict__ b1,
                           const float* __restrict__ W2, const float* __restrict__ b2,
                           float* __restrict__ newea) {
    int t = blockIdx.z;
    long e = (long)blockIdx.x * (blockDim.x >> 5) + (threadIdx.x >> 5);
    if (e >= Ee) return;
    int lane = threadIdx.x & 31;
    int src = EI[((long)t * 2 + 0) * Ee + e];
    int dst = EI[((long)t * 2 + 1) * Ee + e];
    const float* ea = EA + ((long)t * Ee + e) * 15;
    float h = xa[(long)src * EHh + lane] + xb[(long)dst * EHh + lane] + b1[lane];
#pragma unroll
    for (int r = 0; r < 15; ++r) h += ea[r] * W1[(640 + r) * EHh + lane];
    h = (h > 0.f) ? h : 0.01f * h;                  // leaky_relu
    float o = b2[lane];
#pragma unroll
    for (int j = 0; j < 32; ++j) {
        float hj = __shfl(h, j, 32);
        o += hj * W2[j * EHh + lane];
    }
    newea[((long)t * Ee + e) * EHh + lane] = o;
}

// ---------------- attention pass 1: alpha + segment max ----------------
__global__ void edge_alpha_k(const float* __restrict__ Q, const float* __restrict__ Kf,
                             const float* __restrict__ Et, const int* __restrict__ EI, int t,
                             float* __restrict__ alpha, float* __restrict__ amax) {
    long e = (long)blockIdx.x * (blockDim.x >> 5) + (threadIdx.x >> 5);
    if (e >= Ee) return;
    int lane = threadIdx.x & 31;
    int src = EI[((long)t * 2 + 0) * Ee + e];
    int dst = EI[((long)t * 2 + 1) * Ee + e];
    float s = 0.f;
    for (int d = lane; d < Dd; d += 32)
        s += Q[(long)dst * Dd + d] * (Kf[(long)src * Dd + d] + Et[(long)e * Dd + d]);
#pragma unroll
    for (int o = 16; o; o >>= 1) s += __shfl_xor(s, o, 32);
    s *= rsqrtf((float)Dd);
    if (lane == 0) { alpha[e] = s; atomicMaxF(&amax[dst], s); }
}

// ---------------- attention pass 2: exp + segment sum ----------------
__global__ void edge_soft_k(const float* __restrict__ alpha, const int* __restrict__ EI, int t,
                            const float* __restrict__ amax, float* __restrict__ exv,
                            float* __restrict__ denom) {
    long e = (long)blockIdx.x * blockDim.x + threadIdx.x;
    if (e >= Ee) return;
    int dst = EI[((long)t * 2 + 1) * Ee + e];
    float m = amax[dst];
    if (m <= -3.0e38f) m = 0.f;                      // isfinite guard
    float ex = __expf(alpha[e] - m);
    exv[e] = ex;
    atomicAdd(&denom[dst], ex);
}

// ---------------- attention pass 3: weighted scatter of V+E ----------------
__global__ void edge_scatter_k(const float* __restrict__ V, const float* __restrict__ Et,
                               const int* __restrict__ EI, int t,
                               const float* __restrict__ exv, const float* __restrict__ denom,
                               float* __restrict__ acc) {
    long e = (long)blockIdx.x * (blockDim.x >> 5) + (threadIdx.x >> 5);
    if (e >= Ee) return;
    int lane = threadIdx.x & 31;
    int src = EI[((long)t * 2 + 0) * Ee + e];
    int dst = EI[((long)t * 2 + 1) * Ee + e];
    float w = exv[e] / (denom[dst] + 1e-16f);
    for (int d = lane; d < Dd; d += 32)
        atomicAdd(&acc[(long)dst * Dd + d], w * (V[(long)src * Dd + d] + Et[(long)e * Dd + d]));
}

// ---------------- x_ = leaky_relu(acc / 9), in place ----------------
__global__ void act_k(float* p, long n) {
    long i = (long)blockIdx.x * blockDim.x + threadIdx.x;
    if (i < n) {
        float v = p[i] * (1.f / 9.f);
        p[i] = (v > 0.f) ? v : 0.01f * v;
    }
}

// ---------------- graph layernorm reductions ----------------
__global__ void gsum_k(const float* __restrict__ xo, const int* __restrict__ batch,
                       float* __restrict__ gsum) {
    long node = (long)blockIdx.x * (blockDim.x >> 5) + (threadIdx.x >> 5);
    if (node >= Nn) return;
    int lane = threadIdx.x & 31;
    float s = 0.f;
    for (int d = lane; d < Dd; d += 32) s += xo[node * Dd + d];
#pragma unroll
    for (int o = 16; o; o >>= 1) s += __shfl_xor(s, o, 32);
    if (lane == 0) atomicAdd(&gsum[batch[node]], s);
}
__global__ void gvar_k(const float* __restrict__ xo, const int* __restrict__ batch,
                       const float* __restrict__ gmean, float* __restrict__ gvar) {
    long node = (long)blockIdx.x * (blockDim.x >> 5) + (threadIdx.x >> 5);
    if (node >= Nn) return;
    int lane = threadIdx.x & 31;
    float m = gmean[batch[node]];
    float s = 0.f;
    for (int d = lane; d < Dd; d += 32) {
        float c = xo[node * Dd + d] - m;
        s += c * c;
    }
#pragma unroll
    for (int o = 16; o; o >>= 1) s += __shfl_xor(s, o, 32);
    if (lane == 0) atomicAdd(&gvar[batch[node]], s);
}
__global__ void gmeanfin_k(float* g, const float* cnt) {
    int i = blockIdx.x * blockDim.x + threadIdx.x;
    if (i < Gg) g[i] /= fmaxf(cnt[i], 1.f) * (float)Dd;
}
__global__ void gvarfin_k(float* g, const float* cnt) {
    int i = blockIdx.x * blockDim.x + threadIdx.x;
    if (i < Gg) {
        float v = g[i] / (fmaxf(cnt[i], 1.f) * (float)Dd);
        g[i] = rsqrtf(v + 1e-5f);                    // inverse std
    }
}

// ---------------- h = 0.5*(h + norm(x_)*gamma+beta) ----------------
__global__ void hupd_k(float* __restrict__ h, const float* __restrict__ xo,
                       const int* __restrict__ batch, const float* __restrict__ gmean,
                       const float* __restrict__ gistd, const float* __restrict__ gamma,
                       const float* __restrict__ beta) {
    long i = (long)blockIdx.x * blockDim.x + threadIdx.x;
    if (i >= (long)Nn * Dd) return;
    int node = (int)(i / Dd), d = (int)(i % Dd);
    int g = batch[node];
    float xn = (xo[i] - gmean[g]) * gistd[g] * gamma[d] + beta[d];
    h[i] = 0.5f * (h[i] + xn);
}

// ---------------- global max pool ----------------
__global__ void pool_k(const float* __restrict__ h, const int* __restrict__ batch,
                       float* __restrict__ out) {
    long i = (long)blockIdx.x * blockDim.x + threadIdx.x;
    if (i >= (long)Nn * Dd) return;
    int node = (int)(i / Dd), d = (int)(i % Dd);
    atomicMaxF(&out[(long)batch[node] * Dd + d], h[i]);
}

extern "C" void kernel_launch(void* const* d_in, const int* in_sizes, int n_in,
                              void* d_out, int out_size, void* d_ws, size_t ws_size,
                              hipStream_t stream) {
    const float* x    = (const float*)d_in[0];
    const int*   batch= (const int*)  d_in[1];
    const int*   EI   = (const int*)  d_in[2];
    const float* EA   = (const float*)d_in[3];
    const float* W1   = (const float*)d_in[4];
    const float* b1   = (const float*)d_in[5];
    const float* W2   = (const float*)d_in[6];
    const float* b2   = (const float*)d_in[7];
    const float* Wq   = (const float*)d_in[8];
    const float* bq   = (const float*)d_in[9];
    const float* Wk   = (const float*)d_in[10];
    const float* bk   = (const float*)d_in[11];
    const float* Wv   = (const float*)d_in[12];
    const float* bv   = (const float*)d_in[13];
    const float* We   = (const float*)d_in[14];
    const float* be   = (const float*)d_in[15];
    const float* Ws   = (const float*)d_in[16];
    const float* bs   = (const float*)d_in[17];
    const float* gamma= (const float*)d_in[18];
    const float* beta = (const float*)d_in[19];
    float* out = (float*)d_out;

    // workspace carve (floats)
    float* ws   = (float*)d_ws;
    float* h    = ws;  ws += (long)Nn * Dd;
    float* acc  = ws;  ws += (long)Nn * Dd;
    float* qb   = ws;  ws += (long)Nn * Dd;
    float* kb   = ws;  ws += (long)Nn * Dd;
    float* vb   = ws;  ws += (long)Nn * Dd;
    float* etb  = ws;  ws += (long)Ee * Dd;
    float* nea  = ws;  ws += (long)NTt * Ee * EHh;
    float* xa   = ws;  ws += (long)Nn * EHh;
    float* xb   = ws;  ws += (long)Nn * EHh;
    float* alp  = ws;  ws += Ee;
    float* exv  = ws;  ws += Ee;
    float* amax = ws;  ws += Nn;
    float* den  = ws;  ws += Nn;
    float* cnt  = ws;  ws += Gg;
    float* gmn  = ws;  ws += Gg;
    float* gvr  = ws;  ws += Gg;

    const long ND = (long)Nn * Dd;
    auto gemm = [&](const float* A, const float* B, const float* bias, float* C,
                    int M, int K, int Ncol, int accum) {
        dim3 grid((unsigned)((M + BM - 1) / BM), (unsigned)((Ncol + BN - 1) / BN), 1);
        gemm_wmma_k<<<grid, dim3(128, 1, 1), 0, stream>>>(A, B, bias, C, M, K, Ncol, accum);
    };
    auto ewgrid = [&](void) { return dim3((unsigned)((Ee + 7) / 8), 1, 1); };   // wave/edge
    auto nwgrid = dim3((unsigned)((Nn + 7) / 8), 1, 1);                          // wave/node

    // ---- preprocessing ----
    copy_k<<<dim3((ND + 255) / 256), 256, 0, stream>>>(h, x, ND);
    fill_k<<<1, 256, 0, stream>>>(cnt, 0.f, Gg);
    count_k<<<dim3((Nn + 255) / 256), 256, 0, stream>>>(batch, cnt);
    // Xa = x @ W1[0:320,:], Xb = x @ W1[320:640,:]   (b1 added in edge MLP)
    gemm(x, W1, nullptr, xa, Nn, Dd, EHh, 0);
    gemm(x, W1 + 320 * EHh, nullptr, xb, Nn, Dd, EHh, 0);
    edge_mlp_k<<<dim3((unsigned)((Ee + 7) / 8), 1, NTt), 256, 0, stream>>>(
        xa, xb, EI, EA, W1, b1, W2, b2, nea);

    // ---- conv blocks ----
    for (int b = 0; b < NBb; ++b) {
        fill_k<<<dim3((ND + 255) / 256), 256, 0, stream>>>(acc, 0.f, ND);
        for (int t = 0; t < NTt; ++t) {
            long w = (long)(b * NTt + t);
            // root/skip term accumulated directly: acc += h @ Ws + bs
            gemm(h, Ws + w * Dd * Dd, bs + w * Dd, acc, Nn, Dd, Dd, 1);
        }
        for (int t = 0; t < NTt; ++t) {
            long w = (long)(b * NTt + t);
            gemm(h, Wq + w * Dd * Dd, bq + w * Dd, qb, Nn, Dd, Dd, 0);
            gemm(h, Wk + w * Dd * Dd, bk + w * Dd, kb, Nn, Dd, Dd, 0);
            gemm(h, Wv + w * Dd * Dd, bv + w * Dd, vb, Nn, Dd, Dd, 0);
            gemm(nea + (long)t * Ee * EHh, We + w * EHh * Dd, be + w * Dd, etb,
                 Ee, EHh, Dd, 0);
            fill2_k<<<dim3((Nn + 255) / 256), 256, 0, stream>>>(amax, -3.4e38f, den, 0.f, Nn);
            edge_alpha_k<<<ewgrid(), 256, 0, stream>>>(qb, kb, etb, EI, t, alp, amax);
            edge_soft_k<<<dim3((Ee + 255) / 256), 256, 0, stream>>>(alp, EI, t, amax, exv, den);
            edge_scatter_k<<<ewgrid(), 256, 0, stream>>>(vb, etb, EI, t, exv, den, acc);
        }
        act_k<<<dim3((ND + 255) / 256), 256, 0, stream>>>(acc, ND);
        fill2_k<<<1, 256, 0, stream>>>(gmn, 0.f, gvr, 0.f, Gg);
        gsum_k<<<nwgrid, 256, 0, stream>>>(acc, batch, gmn);
        gmeanfin_k<<<1, 256, 0, stream>>>(gmn, cnt);
        gvar_k<<<nwgrid, 256, 0, stream>>>(acc, batch, gmn, gvr);
        gvarfin_k<<<1, 256, 0, stream>>>(gvr, cnt);
        hupd_k<<<dim3((ND + 255) / 256), 256, 0, stream>>>(
            h, acc, batch, gmn, gvr, gamma + (long)b * Dd, beta + (long)b * Dd);
    }

    // ---- global max pool ----
    fill_k<<<dim3(((long)Gg * Dd + 255) / 256), 256, 0, stream>>>(out, -3.4e38f, (long)Gg * Dd);
    pool_k<<<dim3((ND + 255) / 256), 256, 0, stream>>>(h, batch, out);
}
